// GatingLayer1_59648505807079
// MI455X (gfx1250) — matrix operand order repaired and checked by reading, not compiled, not run on hardware
//
#include <hip/hip_runtime.h>

typedef __attribute__((ext_vector_type(16))) _Float16 v16h;
typedef __attribute__((ext_vector_type(8)))  float    v8f;

#define SEQ 1024
#define BSZ 16
#define HID 1024
#define BH  (BSZ * HID)   // 16384

// ---------------------------------------------------------------------------
// Stage 1: partial column sums of x[S,B,H] over chunks of 64 sequence steps.
// grid = (64, 16), block = 256. idx = b*H + h (flat), fully coalesced.
// ---------------------------------------------------------------------------
__global__ void reduce_partial(const float* __restrict__ x,
                               float* __restrict__ partial) {
    const int idx   = blockIdx.x * blockDim.x + threadIdx.x;  // 0..16383
    const int chunk = blockIdx.y;                             // 0..15
    const float* p = x + (size_t)chunk * 64 * BH + idx;
    float s = 0.0f;
#pragma unroll 8
    for (int i = 0; i < 64; ++i) s += p[(size_t)i * BH];
    partial[chunk * BH + idx] = s;
}

// ---------------------------------------------------------------------------
// Stage 2: collapse the 16 partials -> xmean[b,h], scaled by 1/S.
// Deterministic reduction order (no atomics) so timed replays are bit-stable.
// ---------------------------------------------------------------------------
__global__ void reduce_final(const float* __restrict__ partial,
                             float* __restrict__ xmean) {
    const int idx = blockIdx.x * blockDim.x + threadIdx.x;
    float s = 0.0f;
#pragma unroll
    for (int c = 0; c < 16; ++c) s += partial[c * BH + idx];
    xmean[idx] = s * (1.0f / SEQ);
}

// ---------------------------------------------------------------------------
// Stage 3: out[b,h] = xmean[b,h] + (xmean @ Wv^T)[b,h] + bv[h]
// M=16 (one WMMA tile row), N=1024 (64 tiles), K=1024 (32 steps of 32).
// One wave per 16x16 output tile; 8 blocks x 256 threads = 64 waves.
// Fragment layouts per CDNA5 ISA 7.12.2 (wave32):
//   A 16x32 f16 : lane L -> row M=L%16; g=L/16 selects K pattern
//                 VGPR v -> K = (v<4?0:16) + g*8 + (v&3)*2  (+0,+1 packed)
//   B 32x16 f16 : lane L -> col N=L%16; g=L/16 selects K half
//                 VGPR v -> K = (v<4?0:16) + g*8 + (v&3)*2  (+0,+1 packed)
//   C/D 16x16 f32: VGPR r -> M = r + 8*g, N = L%16
// Both A and B fragment pairs are K-contiguous in memory -> float2 loads.
// ---------------------------------------------------------------------------
__global__ void gemm_wmma(const float* __restrict__ xmean,
                          const float* __restrict__ Wv,
                          const float* __restrict__ bv,
                          float* __restrict__ out) {
    const int lane  = threadIdx.x & 31;
    const int wave  = threadIdx.x >> 5;
    const int tileN = blockIdx.x * (blockDim.x >> 5) + wave;  // 0..63
    const int g     = lane >> 4;
    const int mn    = lane & 15;

    const float* __restrict__ arow = xmean + mn * HID;
    const float* __restrict__ brow = Wv + (size_t)(tileN * 16 + mn) * HID;

    v8f acc = {};
    for (int k0 = 0; k0 < HID; k0 += 32) {
        v16h a, b;
#pragma unroll
        for (int v = 0; v < 8; ++v) {
            const int kb = ((v < 4) ? 0 : 16) + g * 8 + (v & 3) * 2;
            const float2 f = *(const float2*)(arow + k0 + kb);
            a[2 * v]     = (_Float16)f.x;
            a[2 * v + 1] = (_Float16)f.y;
        }
#pragma unroll
        for (int v = 0; v < 8; ++v) {
            const int kb = ((v < 4) ? 0 : 16) + g * 8 + (v & 3) * 2;
            const float2 f = *(const float2*)(brow + k0 + kb);
            b[2 * v]     = (_Float16)f.x;
            b[2 * v + 1] = (_Float16)f.y;
        }
        // (neg_a, A, neg_b, B, c_mod, C, reuse_a, reuse_b)
        acc = __builtin_amdgcn_wmma_f32_16x16x32_f16(
            false, a, false, b, (short)0, acc, false, false);
    }

    const int n = tileN * 16 + mn;
#pragma unroll
    for (int r = 0; r < 8; ++r) {
        const int m = r + 8 * g;
        out[m * HID + n] = acc[r] + xmean[m * HID + n] + bv[n];
    }
}

// ---------------------------------------------------------------------------
// Host-side launcher. Inputs: x, Wq, bq, Wk, bk, Wv, bv (Wq/bq/Wk/bk unused:
// softmax over the query axis makes context.mean(q) == V.mean(k) exactly).
// ---------------------------------------------------------------------------
extern "C" void kernel_launch(void* const* d_in, const int* in_sizes, int n_in,
                              void* d_out, int out_size, void* d_ws, size_t ws_size,
                              hipStream_t stream) {
    const float* x  = (const float*)d_in[0];
    const float* Wv = (const float*)d_in[5];
    const float* bv = (const float*)d_in[6];
    float* out = (float*)d_out;

    float* partial = (float*)d_ws;          // 16 * 16384 floats = 1 MB
    float* xmean   = partial + 16 * BH;     // 16384 floats = 64 KB

    reduce_partial<<<dim3(64, 16), 256, 0, stream>>>(x, partial);
    reduce_final<<<64, 256, 0, stream>>>(partial, xmean);
    gemm_wmma<<<8, 256, 0, stream>>>(xmean, Wv, bv, out);
}
